// VQ_VAE_codebook_loss_41729902248238
// MI455X (gfx1250) — matrix operand order repaired and checked by reading, not compiled, not run on hardware
//
#include <hip/hip_runtime.h>

#define NB   32
#define CCH  64
#define PPX  4096
#define SCB  512
#define IDX_OFF 2
#define OUT_OFF (2 + NB*PPX)
#define TOTAL_ELEMS (NB*CCH*PPX)

typedef __attribute__((ext_vector_type(16))) __bf16 v16bf;
typedef __attribute__((ext_vector_type(8)))  float  v8f;

struct BPack { uint4 lo, hi; };   // 32 bytes == v16bf

__device__ __forceinline__ unsigned short f2bf(float f) {
  unsigned u = __float_as_uint(f);
  u += 0x7FFFu + ((u >> 16) & 1u);          // round-to-nearest-even
  return (unsigned short)(u >> 16);
}

// -------- pass 0: codebook -> bf16 (ws) + exact fp32 row norms --------
__global__ void vq_prep(const float* __restrict__ cb,
                        unsigned short* __restrict__ cb_bf,
                        float* __restrict__ c2) {
  int r = blockIdx.x * blockDim.x + threadIdx.x;
  if (r >= SCB) return;
  const float* row = cb + r * CCH;
  float s = 0.f;
  #pragma unroll 8
  for (int c = 0; c < CCH; ++c) {
    float v = row[c];
    s = fmaf(v, v, s);
    cb_bf[r * CCH + c] = f2bf(v);
  }
  c2[r] = s;
}

// -------- pass 1: WMMA distance GEMM + argmin --------
// block = 128 threads (4 waves); wave owns 32 pixels (two 16-wide tiles)
__global__ void __launch_bounds__(128)
vq_argmin(const float* __restrict__ x,
          const unsigned short* __restrict__ cb_bf,
          const float* __restrict__ c2g,
          float* __restrict__ outf) {
  __shared__ float c2s[SCB];
  const int tid = threadIdx.x;
  for (int k = tid; k < SCB; k += 128) c2s[k] = c2g[k];
  __syncthreads();

  const int n    = blockIdx.x >> 5;           // 32 blocks per image
  const int pblk = (blockIdx.x & 31) * 128;
  const int w    = tid >> 5;
  const int lane = tid & 31;
  const int lo   = lane & 15;
  const int hi   = lane >> 4;

  const float* xn = x + (size_t)n * CCH * PPX;

  // B operands (loop-invariant): e[i] <-> K = i + 8*hi (+8 if i>=8), col = lo
  union VB { v16bf v; unsigned short u[16]; };
  VB b0[2], b1[2];
  #pragma unroll
  for (int t = 0; t < 2; ++t) {
    const int p = pblk + w * 32 + t * 16 + lo;
    #pragma unroll
    for (int i = 0; i < 16; ++i) {
      const int c0 = i + 8 * hi + (i >= 8 ? 8 : 0);
      b0[t].u[i] = f2bf(xn[(c0 << 12) + p]);          // channels 0..31
      b1[t].u[i] = f2bf(xn[((c0 + 32) << 12) + p]);   // channels 32..63
    }
  }

  float minv[2] = {3.4e38f, 3.4e38f};
  int   mini[2] = {0, 0};

  for (int sb = 0; sb < SCB; sb += 16) {
    // A operand: row = sb+lo; e[0..7] = K 8*hi.. ; e[8..15] = K 16+8*hi..
    const uint4* rowp = (const uint4*)(cb_bf + (sb + lo) * CCH); // 8 x uint4 per row
    if (sb + 16 < SCB)
      __builtin_prefetch(cb_bf + (sb + 16 + lo) * CCH, 0, 0);   // global_prefetch_b8
    BPack pa0{rowp[hi],     rowp[2 + hi]};   // channels [0,32)
    BPack pa1{rowp[4 + hi], rowp[6 + hi]};   // channels [32,64)
    v16bf a0 = __builtin_bit_cast(v16bf, pa0);
    v16bf a1 = __builtin_bit_cast(v16bf, pa1);

    #pragma unroll
    for (int t = 0; t < 2; ++t) {
      v8f acc = {0.f, 0.f, 0.f, 0.f, 0.f, 0.f, 0.f, 0.f};
      acc = __builtin_amdgcn_wmma_f32_16x16x32_bf16(false, a0, false, b0[t].v,
                                                    (short)0, acc, false, false);
      acc = __builtin_amdgcn_wmma_f32_16x16x32_bf16(false, a1, false, b1[t].v,
                                                    (short)0, acc, false, false);
      // all 8 slots of this lane belong to pixel (base + lo), s = sb + i + 8*hi
      #pragma unroll
      for (int i = 0; i < 8; ++i) {
        const int   s = sb + i + 8 * hi;
        const float d = fmaf(-2.f, acc[i], c2s[s]);   // x^2 invariant -> dropped
        if (d < minv[t]) { minv[t] = d; mini[t] = s; }
      }
    }
  }

  #pragma unroll
  for (int t = 0; t < 2; ++t) {
    const float ov = __shfl_xor(minv[t], 16, 32);
    const int   oi = __shfl_xor(mini[t], 16, 32);
    if (ov < minv[t] || (ov == minv[t] && oi < mini[t])) { minv[t] = ov; mini[t] = oi; }
    if (hi == 0) {
      const int p = pblk + w * 32 + t * 16 + lo;
      outf[IDX_OFF + n * PPX + p] = (float)mini[t];
    }
  }
}

// -------- pass 2: gather x_q, write out, exact (x - x_q)^2 partials --------
__global__ void __launch_bounds__(256)
vq_gather(const float* __restrict__ x,
          const float* __restrict__ cb,
          float* __restrict__ outf,
          float* __restrict__ partials) {
  __shared__ float red[256];
  const int tid = threadIdx.x;
  const int gp  = blockIdx.x * 256 + tid;
  const int n   = gp >> 12;
  const int p   = gp & (PPX - 1);

  int idx = (int)outf[IDX_OFF + gp];
  idx &= (SCB - 1);

  const float4* row4 = (const float4*)(cb + idx * CCH);
  const float*  xn   = x    + (size_t)n * CCH * PPX;
  float*        on   = outf + OUT_OFF + (size_t)n * CCH * PPX;

  float s = 0.f;
  #pragma unroll
  for (int j = 0; j < 16; ++j) {
    float4 q = row4[j];
    const int c = j * 4;
    float xa = xn[((c + 0) << 12) + p];
    float xb = xn[((c + 1) << 12) + p];
    float xc = xn[((c + 2) << 12) + p];
    float xd = xn[((c + 3) << 12) + p];
    on[((c + 0) << 12) + p] = q.x;
    on[((c + 1) << 12) + p] = q.y;
    on[((c + 2) << 12) + p] = q.z;
    on[((c + 3) << 12) + p] = q.w;
    float d0 = xa - q.x, d1 = xb - q.y, d2 = xc - q.z, d3 = xd - q.w;
    s += d0 * d0 + d1 * d1 + d2 * d2 + d3 * d3;
  }
  red[tid] = s;
  __syncthreads();
  for (int st = 128; st > 0; st >>= 1) {
    if (tid < st) red[tid] += red[tid + st];
    __syncthreads();
  }
  if (tid == 0) partials[blockIdx.x] = red[0];
}

// -------- pass 3: deterministic final reduce, both losses --------
__global__ void vq_final(const float* __restrict__ partials,
                         float* __restrict__ outf) {
  const int lane = threadIdx.x;   // 32 threads
  float v = 0.f;
  for (int j = 0; j < 16; ++j) v += partials[lane + 32 * j];
  #pragma unroll
  for (int off = 16; off > 0; off >>= 1) v += __shfl_xor(v, off, 32);
  if (lane == 0) {
    const float m = v * (1.0f / (float)TOTAL_ELEMS);
    outf[0] = m;   // loss_codebook
    outf[1] = m;   // loss_commitment (same forward value)
  }
}

extern "C" void kernel_launch(void* const* d_in, const int* in_sizes, int n_in,
                              void* d_out, int out_size, void* d_ws, size_t ws_size,
                              hipStream_t stream) {
  const float* x  = (const float*)d_in[0];
  const float* cb = (const float*)d_in[1];
  float* outf = (float*)d_out;

  unsigned short* cb_bf = (unsigned short*)d_ws;                       // 64 KB
  float* c2       = (float*)((char*)d_ws + SCB * CCH * sizeof(unsigned short));
  float* partials = (float*)((char*)c2 + SCB * sizeof(float));         // 512 floats

  vq_prep  <<<2,    256, 0, stream>>>(cb, cb_bf, c2);
  vq_argmin<<<1024, 128, 0, stream>>>(x, cb_bf, c2, outf);
  vq_gather<<<512,  256, 0, stream>>>(x, cb, outf, partials);
  vq_final <<<1,    32,  0, stream>>>(partials, outf);
}
